// SuperGlue_6219112645318
// MI455X (gfx1250) — matrix operand order repaired and checked by reading, not compiled, not run on hardware
//
#include <hip/hip_runtime.h>
#include <math.h>

// ---------------------------------------------------------------------------
// SuperGlue forward for MI455X (gfx1250, wave32).
// All dense math runs through v_wmma_f32_16x16x32_f16 (D = A x B^T, f32 acc).
// GEMM inner loop is ping-pong software-pipelined (unroll x2, no register
// rotation): set-A fragments are consumed while set-B loads are in flight,
// so loadcnt waits are partial instead of full drains. Weights pre-converted
// to f16 once. Activations position-major f16 [N][C]; Sinkhorn f32 in L2.
// ---------------------------------------------------------------------------

typedef __attribute__((ext_vector_type(16))) _Float16 v16h;
typedef __attribute__((ext_vector_type(8)))  _Float16 v8h;
typedef __attribute__((ext_vector_type(8)))  float    v8f;

#define NPTS 1024
#define SK_NORM (-7.6246189861593985f)   /* -log(2048) */
#define SK_LOGN (6.9314718055994531f)    /*  log(1024) */

// ------------------------------ reductions ---------------------------------
__device__ inline float block_max(float x, float* red) {
#pragma unroll
  for (int o = 16; o > 0; o >>= 1) x = fmaxf(x, __shfl_xor(x, o));
  if ((threadIdx.x & 31) == 0) red[threadIdx.x >> 5] = x;
  __syncthreads();
  x = fmaxf(fmaxf(fmaxf(red[0], red[1]), fmaxf(red[2], red[3])),
            fmaxf(fmaxf(red[4], red[5]), fmaxf(red[6], red[7])));
  __syncthreads();
  return x;
}
__device__ inline float block_sum(float x, float* red) {
#pragma unroll
  for (int o = 16; o > 0; o >>= 1) x += __shfl_xor(x, o);
  if ((threadIdx.x & 31) == 0) red[threadIdx.x >> 5] = x;
  __syncthreads();
  x = red[0] + red[1] + red[2] + red[3] + red[4] + red[5] + red[6] + red[7];
  __syncthreads();
  return x;
}

// --------------------------- generic WMMA GEMM -----------------------------
// O[r][c] = scale * sum_k A[r][k] * B[c][k]  (+bias, BN, ReLU epilogue)
// A, B: f16 row-major over K. grid = (C/64, 8, nz); block = 256 (8 waves);
// each wave owns a 16-row strip x 64 output cols (4 WMMA tiles, A reused 4x).
// K is always a multiple of 64 here (64/256/512/1024) -> even step count.

// Load one K-step's fragments in contiguous k order, the same for A and B:
//   A (16x32): lane half=0 -> K = 0..15, half=1 -> K = 16..31 (contiguous)
//   B (32x16): lane half=0 -> K = 0..15, half=1 -> K = 16..31 (contiguous)
__device__ inline void load_frags(const _Float16* __restrict__ Ap,
                                  const _Float16* __restrict__ Bp, int ldb,
                                  int c0, int m, int half, int k0,
                                  v16h& a, v16h* b) {
  const _Float16* pa = Ap + k0 + 16 * half;
  a = *(const v16h*)pa;
#pragma unroll
  for (int t = 0; t < 4; ++t)
    b[t] = *(const v16h*)(Bp + (long long)(c0 + t * 16 + m) * ldb + k0 + 16 * half);
}

__global__ __launch_bounds__(256) void gemm_abt(
    const _Float16* __restrict__ A, int lda,
    const _Float16* __restrict__ B, int ldb,
    int K, float scale,
    const float* __restrict__ bias,
    const float* __restrict__ bn_be, const float* __restrict__ bn_g,
    const float* __restrict__ bn_mu, const float* __restrict__ bn_var,
    int relu, int headperm, int colmul, int zccol,
    float* __restrict__ of32, int ldo32, long long zcplane,
    _Float16* __restrict__ of16, int ldo16,
    _Float16* __restrict__ of16T, int ldT,
    long long zA, long long zB)
{
  const int z    = blockIdx.z;
  const int lane = threadIdx.x & 31;
  const int wave = threadIdx.x >> 5;
  const int r0   = (blockIdx.y * 8 + wave) * 16;
  const int c0   = blockIdx.x * 64;
  const int m    = lane & 15;    // A row within tile / B row (=output col) / C col
  const int half = lane >> 4;

  const _Float16* Ap = A + (long long)z * zA + (long long)(r0 + m) * lda;
  const _Float16* Bp = B + (long long)z * zB;

  v8f acc[4] = {};

  // ---- ping-pong pipelined main loop (unroll x2, no register rotation) ----
  v16h aA, bA[4], aB, bB[4];
  load_frags(Ap, Bp, ldb, c0, m, half, 0, aA, bA);
  int k0 = 0;
  while (true) {
    // issue loads for step k0+32 into set B, then consume set A
    load_frags(Ap, Bp, ldb, c0, m, half, k0 + 32, aB, bB);
    if (k0 + 64 < K)  // hint next pair (global_prefetch_b8)
      __builtin_prefetch(Ap + k0 + 64 + 8 * half, 0, 1);
#pragma unroll
    for (int t = 0; t < 4; ++t)
      acc[t] = __builtin_amdgcn_wmma_f32_16x16x32_f16(
          false, aA, false, bA[t], (short)0, acc[t], false, false);

    // issue loads for step k0+64 into set A (if any), then consume set B
    const int k2 = k0 + 64;
    const bool more = k2 < K;
    if (more) load_frags(Ap, Bp, ldb, c0, m, half, k2, aA, bA);
#pragma unroll
    for (int t = 0; t < 4; ++t)
      acc[t] = __builtin_amdgcn_wmma_f32_16x16x32_f16(
          false, aB, false, bB[t], (short)0, acc[t], false, false);
    if (!more) break;
    k0 = k2;
  }

  // ---- epilogue: C layout = lane col (lane&15), rows rr + 8*half ----------
#pragma unroll
  for (int t = 0; t < 4; ++t) {
    const int cc = c0 + t * 16 + m;           // original output channel
    float bv = bias ? bias[cc] : 0.f;
    float bsc = 1.f, bsh = 0.f;
    if (bn_g) {
      float inv = bn_g[cc] * rsqrtf(bn_var[cc] + 1e-5f);
      bsc = inv; bsh = bn_be[cc] - bn_mu[cc] * inv;
    }
    // head-interleave permutation: channel dd*4+h -> h*64+dd
    const int oc = headperm ? (((cc & 3) << 6) | (cc >> 2))
                            : (z * zccol + cc * colmul);
#pragma unroll
    for (int rr = 0; rr < 8; ++rr) {
      const int r = r0 + rr + 8 * half;
      float x = acc[t][rr] * scale + bv;
      if (bn_g) x = x * bsc + bsh;
      if (relu) x = fmaxf(x, 0.f);
      if (of32)  of32[(long long)z * zcplane + (long long)r * ldo32 + cc] = x;
      if (of16)  of16[(long long)r * ldo16 + oc] = (_Float16)x;
      if (of16T) of16T[(long long)oc * ldT + r] = (_Float16)x;
    }
  }
}

// ------------------- weight f32 -> f16 conversion --------------------------
// One layer's 6 matrices packed: p0,p1,p2,merge (4x65536) | m0 (262144) | m1 (131072)
struct W6 { const float* src[6]; };
__global__ __launch_bounds__(256) void conv_w6(W6 w, _Float16* __restrict__ dst) {
  const int i = blockIdx.x * 256 + threadIdx.x;   // 655360 total
  int seg, off;
  if (i < 262144)      { seg = i >> 16; off = i & 65535; }
  else if (i < 524288) { seg = 4; off = i - 262144; }
  else                 { seg = 5; off = i - 524288; }
  dst[i] = (_Float16)w.src[seg][off];
}
__global__ __launch_bounds__(256) void conv_w1(const float* __restrict__ s,
                                               _Float16* __restrict__ d) {
  const int i = blockIdx.x * 256 + threadIdx.x;
  d[i] = (_Float16)s[i];
}

// ------------------- descriptor normalize + transpose ----------------------
__global__ __launch_bounds__(256) void init_desc(const float* __restrict__ din,
                                                 float* __restrict__ desc) {
  __shared__ float red[8];
  const int n = blockIdx.x, c = threadIdx.x;
  float v = din[(size_t)c * NPTS + n];
  float ss = block_sum(v * v, red);
  float nrm = fmaxf(sqrtf(ss), 1e-12f);
  desc[(size_t)n * 256 + c] = v / nrm;
}

// --------------------------- keypoint encoder ------------------------------
struct KencP {
  const float* W[5]; const float* b[5];
  const float* be[4]; const float* g[4]; const float* mu[4]; const float* var[4];
};

__global__ __launch_bounds__(256) void kenc_add(
    const float* __restrict__ kpts, const float* __restrict__ scores,
    const int* __restrict__ hin, const int* __restrict__ win,
    KencP P, float* __restrict__ desc, _Float16* __restrict__ cat) {
  __shared__ float buf0[256], buf1[256];
  const int n = blockIdx.x, t = threadIdx.x;
  const float H = (float)(*hin), W = (float)(*win);
  const float mx = fmaxf(W, H) * 0.7f;
  if (t == 0) {
    buf0[0] = (kpts[n * 2 + 0] - W * 0.5f) / mx;
    buf0[1] = (kpts[n * 2 + 1] - H * 0.5f) / mx;
    buf0[2] = scores[n];
  }
  __syncthreads();
  const int chs[6] = {3, 32, 64, 128, 256, 256};
  float* in = buf0; float* out = buf1;
  for (int L = 0; L < 5; ++L) {
    const int ci = chs[L], co = chs[L + 1];
    if (t < co) {
      const float* w = P.W[L] + (size_t)t * ci;
      float acc = P.b[L][t];
      for (int k = 0; k < ci; ++k) acc += w[k] * in[k];
      if (L < 4) {
        float inv = P.g[L][t] * rsqrtf(P.var[L][t] + 1e-5f);
        acc = (acc - P.mu[L][t]) * inv + P.be[L][t];
        acc = fmaxf(acc, 0.f);
      }
      out[t] = acc;
    }
    __syncthreads();
    float* tmp = in; in = out; out = tmp;
  }
  float d = desc[(size_t)n * 256 + t] + in[t];
  desc[(size_t)n * 256 + t] = d;
  cat[(size_t)n * 512 + t] = (_Float16)d;
}

// ------------------------------- softmax -----------------------------------
// One block per row; each thread owns exactly 4 elements, cached in registers.
__global__ __launch_bounds__(256) void softmax_rows(const float* __restrict__ S,
                                                    _Float16* __restrict__ P) {
  __shared__ float red[8];
  const int row = blockIdx.x, t = threadIdx.x;
  const float* s = S + (size_t)row * NPTS;
  float v0 = s[t], v1 = s[t + 256], v2 = s[t + 512], v3 = s[t + 768];
  float mx = block_max(fmaxf(fmaxf(v0, v1), fmaxf(v2, v3)), red);
  v0 = __expf(v0 - mx); v1 = __expf(v1 - mx);
  v2 = __expf(v2 - mx); v3 = __expf(v3 - mx);
  float sum = block_sum(v0 + v1 + v2 + v3, red);
  const float inv = 1.f / sum;
  _Float16* p = P + (size_t)row * NPTS;
  p[t]       = (_Float16)(v0 * inv);
  p[t + 256] = (_Float16)(v1 * inv);
  p[t + 512] = (_Float16)(v2 * inv);
  p[t + 768] = (_Float16)(v3 * inv);
}

// --------------------------- residual update -------------------------------
__global__ __launch_bounds__(256) void apply_delta(float* __restrict__ desc,
                                                   _Float16* __restrict__ cat,
                                                   const float* __restrict__ delta) {
  const int i = blockIdx.x * 256 + threadIdx.x;   // 1024*256 elems
  const int n = i >> 8, c = i & 255;
  float d = desc[i] + delta[i];
  desc[i] = d;
  cat[(size_t)n * 512 + c] = (_Float16)d;
}

// ------------------------------ Sinkhorn -----------------------------------
__global__ __launch_bounds__(256) void build_couplings(
    const float* __restrict__ S, const float* __restrict__ alpha,
    float* __restrict__ Cp, float* __restrict__ u, float* __restrict__ v) {
  const int i = blockIdx.y;
  const int j = blockIdx.x * 256 + threadIdx.x;
  if (j > NPTS) return;
  float val = (i < NPTS && j < NPTS) ? S[(size_t)i * NPTS + j] : alpha[0];
  Cp[(size_t)i * (NPTS + 1) + j] = val;
  if (i == 0) { u[j] = 0.f; v[j] = 0.f; }
}

__global__ __launch_bounds__(256) void sinkhorn_u(const float* __restrict__ Cp,
                                                  const float* __restrict__ v,
                                                  float* __restrict__ u) {
  __shared__ float red[8];
  const int i = blockIdx.x, t = threadIdx.x;
  const float* row = Cp + (size_t)i * (NPTS + 1);
  float x0 = row[t]       + v[t];
  float x1 = row[t + 256] + v[t + 256];
  float x2 = row[t + 512] + v[t + 512];
  float x3 = row[t + 768] + v[t + 768];
  float x4 = (t == 0) ? (row[1024] + v[1024]) : -3.4e38f;
  float mx = block_max(fmaxf(fmaxf(fmaxf(x0, x1), fmaxf(x2, x3)), x4), red);
  float s = __expf(x0 - mx) + __expf(x1 - mx) + __expf(x2 - mx) + __expf(x3 - mx);
  if (t == 0) s += __expf(x4 - mx);
  s = block_sum(s, red);
  if (t == 0) {
    float lmu = (i < NPTS) ? SK_NORM : (SK_LOGN + SK_NORM);
    u[i] = lmu - (mx + __logf(s));
  }
}

__global__ __launch_bounds__(256) void sinkhorn_v(const float* __restrict__ Cp,
                                                  const float* __restrict__ u,
                                                  float* __restrict__ v) {
  __shared__ float red[8];
  const int j = blockIdx.x, t = threadIdx.x;
  const float* col = Cp + j;
  float x0 = col[(size_t)(t)       * (NPTS + 1)] + u[t];
  float x1 = col[(size_t)(t + 256) * (NPTS + 1)] + u[t + 256];
  float x2 = col[(size_t)(t + 512) * (NPTS + 1)] + u[t + 512];
  float x3 = col[(size_t)(t + 768) * (NPTS + 1)] + u[t + 768];
  float x4 = (t == 0) ? (col[(size_t)1024 * (NPTS + 1)] + u[1024]) : -3.4e38f;
  float mx = block_max(fmaxf(fmaxf(fmaxf(x0, x1), fmaxf(x2, x3)), x4), red);
  float s = __expf(x0 - mx) + __expf(x1 - mx) + __expf(x2 - mx) + __expf(x3 - mx);
  if (t == 0) s += __expf(x4 - mx);
  s = block_sum(s, red);
  if (t == 0) {
    float lnu = (j < NPTS) ? SK_NORM : (SK_LOGN + SK_NORM);
    v[j] = lnu - (mx + __logf(s));
  }
}

__global__ void finalize_ot(const float* __restrict__ Cp, const float* __restrict__ u,
                            const float* __restrict__ v, float* __restrict__ out) {
  long long idx = (long long)blockIdx.x * 256 + threadIdx.x;
  const long long total = (long long)(NPTS + 1) * (NPTS + 1);
  if (idx >= total) return;
  int i = (int)(idx / (NPTS + 1)), j = (int)(idx % (NPTS + 1));
  out[idx] = Cp[idx] + u[i] + v[j] - SK_NORM;
}

// ------------------------------ host side ----------------------------------
static inline void gemm(hipStream_t st,
    const _Float16* A, int lda, const _Float16* B, int ldb, int K, int C, int nz,
    float scale, const float* bias,
    const float* be, const float* g, const float* mu, const float* var, int relu,
    int headperm, int colmul, int zccol,
    float* of32, int ldo32, long long zcplane,
    _Float16* of16, int ldo16, _Float16* of16T, int ldT,
    long long zA, long long zB)
{
  dim3 gr(C / 64, 8, nz), bl(256);
  gemm_abt<<<gr, bl, 0, st>>>(A, lda, B, ldb, K, scale, bias, be, g, mu, var,
      relu, headperm, colmul, zccol, of32, ldo32, zcplane, of16, ldo16, of16T, ldT, zA, zB);
}

extern "C" void kernel_launch(void* const* d_in, const int* in_sizes, int n_in,
                              void* d_out, int out_size, void* d_ws, size_t ws_size,
                              hipStream_t stream) {
  (void)in_sizes; (void)n_in; (void)out_size; (void)ws_size;
  auto F = [&](int i) { return (const float*)d_in[i]; };

  // ---- params pytree layout (alphabetical keys, lists in order) ----
  // d_in: 0..5 tensors, 6=bin_score, 7=final.W, 8=final.b,
  // gnn layer L base = 9+16L: mergeW,mergeb, mlp0W,mlp0b, bn{be,g,mu,var},
  //                           mlp1W,mlp1b, p0W,p0b, p1W,p1b, p2W,p2b
  // kenc base = 9+288 = 297, height = 323, width = 324
  const float* bin  = F(6);
  const float* finW = F(7);
  const float* finB = F(8);
  const int kb0 = 9 + 18 * 16;
  const int* hptr = (const int*)d_in[kb0 + 26];
  const int* wptr = (const int*)d_in[kb0 + 27];

  // ---- workspace carve-up (~45 MB) ----
  size_t off = 0;
  auto alloc = [&](size_t bytes) -> char* {
    char* p = (char*)d_ws + off;
    off = (off + bytes + 255) & ~(size_t)255;
    return p;
  };
  float*    desc0  = (float*)alloc(NPTS * 256 * 4);
  float*    desc1  = (float*)alloc(NPTS * 256 * 4);
  _Float16* cat0   = (_Float16*)alloc(NPTS * 512 * 2);
  _Float16* cat1   = (_Float16*)alloc(NPTS * 512 * 2);
  _Float16* qbuf   = (_Float16*)alloc(NPTS * 256 * 2);
  _Float16* kbuf   = (_Float16*)alloc(NPTS * 256 * 2);
  _Float16* vT     = (_Float16*)alloc(256 * NPTS * 2);
  float*    att    = (float*)alloc((size_t)4 * NPTS * NPTS * 4);
  _Float16* prob   = (_Float16*)alloc((size_t)4 * NPTS * NPTS * 2);
  _Float16* msg    = (_Float16*)alloc(NPTS * 256 * 2);
  _Float16* h1     = (_Float16*)alloc(NPTS * 512 * 2);
  float*    delta0 = (float*)alloc(NPTS * 256 * 4);
  float*    delta1 = (float*)alloc(NPTS * 256 * 4);
  _Float16* md0    = (_Float16*)alloc(NPTS * 256 * 2);
  _Float16* md1    = (_Float16*)alloc(NPTS * 256 * 2);
  float*    S      = (float*)alloc((size_t)NPTS * NPTS * 4);
  float*    Cp     = (float*)alloc((size_t)(NPTS + 1) * (NPTS + 1) * 4);
  float*    uvec   = (float*)alloc((NPTS + 1) * 4);
  float*    vvec   = (float*)alloc((NPTS + 1) * 4);
  _Float16* wh     = (_Float16*)alloc((size_t)655360 * 2);  // per-layer f16 weights
  _Float16* whfin  = (_Float16*)alloc((size_t)65536 * 2);   // final.W f16
  // f16 weight scratch layout (halves):
  const int WO_P0 = 0, WO_P1 = 65536, WO_P2 = 131072, WO_MG = 196608,
            WO_M0 = 262144, WO_M1 = 524288;

  // ---- front-end: normalize descriptors, keypoint encoder ----
  init_desc<<<NPTS, 256, 0, stream>>>(F(0), desc0);
  init_desc<<<NPTS, 256, 0, stream>>>(F(1), desc1);
  KencP kp;
  for (int l = 0; l < 4; ++l) {
    kp.W[l]  = F(kb0 + 6 * l + 0); kp.b[l]   = F(kb0 + 6 * l + 1);
    kp.be[l] = F(kb0 + 6 * l + 2); kp.g[l]   = F(kb0 + 6 * l + 3);
    kp.mu[l] = F(kb0 + 6 * l + 4); kp.var[l] = F(kb0 + 6 * l + 5);
  }
  kp.W[4] = F(kb0 + 24); kp.b[4] = F(kb0 + 25);
  kenc_add<<<NPTS, 256, 0, stream>>>(F(2), F(4), hptr, wptr, kp, desc0, cat0);
  kenc_add<<<NPTS, 256, 0, stream>>>(F(3), F(5), hptr, wptr, kp, desc1, cat1);

  // ---- 18 GNN layers ----
  const long long PLANE = (long long)NPTS * NPTS;
  for (int L = 0; L < 18; ++L) {
    const int gb = 9 + 16 * L;
    const float *mergeb = F(gb + 1);
    const float *m0b = F(gb + 3);
    const float *bnbe = F(gb + 4), *bng = F(gb + 5), *bnmu = F(gb + 6), *bnvar = F(gb + 7);
    const float *m1b = F(gb + 9);
    const float *p0b = F(gb + 11), *p1b = F(gb + 13), *p2b = F(gb + 15);
    const bool cross = (L & 1);

    // convert this layer's 6 weight matrices to f16 (read once, reused by all tiles)
    W6 w6;
    w6.src[0] = F(gb + 10); w6.src[1] = F(gb + 12); w6.src[2] = F(gb + 14);
    w6.src[3] = F(gb + 0);  w6.src[4] = F(gb + 2);  w6.src[5] = F(gb + 8);
    conv_w6<<<655360 / 256, 256, 0, stream>>>(w6, wh);

    for (int s = 0; s < 2; ++s) {
      _Float16* catx   = s ? cat1 : cat0;
      _Float16* catsrc = cross ? (s ? cat0 : cat1) : catx;
      float*    dlt    = s ? delta1 : delta0;

      // q/k/v projections (head-permuted f16 outputs; v stored transposed)
      gemm(stream, catx,   512, wh + WO_P0, 256, 256, 256, 1, 1.f, p0b,
           nullptr, nullptr, nullptr, nullptr, 0, 1, 1, 0,
           nullptr, 0, 0, qbuf, 256, nullptr, 0, 0, 0);
      gemm(stream, catsrc, 512, wh + WO_P1, 256, 256, 256, 1, 1.f, p1b,
           nullptr, nullptr, nullptr, nullptr, 0, 1, 1, 0,
           nullptr, 0, 0, kbuf, 256, nullptr, 0, 0, 0);
      gemm(stream, catsrc, 512, wh + WO_P2, 256, 256, 256, 1, 1.f, p2b,
           nullptr, nullptr, nullptr, nullptr, 0, 1, 1, 0,
           nullptr, 0, 0, nullptr, 0, vT, 1024, 0, 0);

      // attention scores (batched over heads), softmax, message
      gemm(stream, qbuf, 256, kbuf, 256, 64, 1024, 4, 0.125f, nullptr,
           nullptr, nullptr, nullptr, nullptr, 0, 0, 1, 0,
           att, 1024, PLANE, nullptr, 0, nullptr, 0, 64, 64);
      softmax_rows<<<4 * NPTS, 256, 0, stream>>>(att, prob);
      gemm(stream, prob, 1024, vT, 1024, 1024, 64, 4, 1.f, nullptr,
           nullptr, nullptr, nullptr, nullptr, 0, 0, 4, 1,
           nullptr, 0, 0, msg, 256, nullptr, 0, PLANE, 64 * 1024);

      // merge -> cat[:,256:512]
      gemm(stream, msg, 256, wh + WO_MG, 256, 256, 256, 1, 1.f, mergeb,
           nullptr, nullptr, nullptr, nullptr, 0, 0, 1, 0,
           nullptr, 0, 0, catx + 256, 512, nullptr, 0, 0, 0);

      // MLP: 512 -> 512 (BN+ReLU) -> 256 (delta)
      gemm(stream, catx, 512, wh + WO_M0, 512, 512, 512, 1, 1.f, m0b,
           bnbe, bng, bnmu, bnvar, 1, 0, 1, 0,
           nullptr, 0, 0, h1, 512, nullptr, 0, 0, 0);
      gemm(stream, h1, 512, wh + WO_M1, 512, 512, 256, 1, 1.f, m1b,
           nullptr, nullptr, nullptr, nullptr, 0, 0, 1, 0,
           dlt, 256, 0, nullptr, 0, nullptr, 0, 0, 0);
    }
    apply_delta<<<NPTS, 256, 0, stream>>>(desc0, cat0, delta0);
    apply_delta<<<NPTS, 256, 0, stream>>>(desc1, cat1, delta1);
  }

  // ---- final projection + match scores ----
  conv_w1<<<65536 / 256, 256, 0, stream>>>(finW, whfin);
  gemm(stream, cat0, 512, whfin, 256, 256, 256, 1, 1.f, finB,
       nullptr, nullptr, nullptr, nullptr, 0, 0, 1, 0,
       nullptr, 0, 0, md0, 256, nullptr, 0, 0, 0);
  gemm(stream, cat1, 512, whfin, 256, 256, 256, 1, 1.f, finB,
       nullptr, nullptr, nullptr, nullptr, 0, 0, 1, 0,
       nullptr, 0, 0, md1, 256, nullptr, 0, 0, 0);
  gemm(stream, md0, 256, md1, 256, 256, 1024, 1, 1.f, nullptr,
       nullptr, nullptr, nullptr, nullptr, 0, 0, 1, 0,
       S, 1024, 0, nullptr, 0, nullptr, 0, 0, 0);

  // ---- log-Sinkhorn (100 iterations) + output ----
  build_couplings<<<dim3(5, NPTS + 1), 256, 0, stream>>>(S, bin, Cp, uvec, vvec);
  for (int it = 0; it < 100; ++it) {
    sinkhorn_u<<<NPTS + 1, 256, 0, stream>>>(Cp, vvec, uvec);
    sinkhorn_v<<<NPTS + 1, 256, 0, stream>>>(Cp, uvec, vvec);
  }
  const long long total = (long long)(NPTS + 1) * (NPTS + 1);
  finalize_ot<<<(int)((total + 255) / 256), 256, 0, stream>>>(Cp, uvec, vvec, (float*)d_out);
}
